// GIN_652835029484
// MI455X (gfx1250) — matrix operand order repaired and checked by reading, not compiled, not run on hardware
//
#include <hip/hip_runtime.h>

typedef __attribute__((ext_vector_type(16))) __bf16 v16bf;
typedef __attribute__((ext_vector_type(8)))  float  v8f;

#define N_NODES 100000
#define N_EDGES 1600000
#define HID 128

__device__ __forceinline__ unsigned short f32_bf16(float f) {
    unsigned int u = __float_as_uint(f);
    u += 0x7FFFu + ((u >> 16) & 1u);   // round-to-nearest-even
    return (unsigned short)(u >> 16);
}

// Pack W[K x 128] (f32 row-major) into per-lane WMMA B fragments (bf16).
// dst[((kt*8 + ct)*32 + lane)*16 + j] = W[kt*32 + (lane>>4)*8 + (j&7) + (j>>3)*16][ct*16 + (lane&15)]
__global__ void pack_w_kernel(const float* __restrict__ W, unsigned short* __restrict__ dst,
                              int Kin, int nkt) {
    int i = blockIdx.x * blockDim.x + threadIdx.x;
    int total = nkt << 12;           // nkt*8*32*16
    if (i >= total) return;
    int j    = i & 15;
    int lane = (i >> 4) & 31;
    int ct   = (i >> 9) & 7;
    int kt   = i >> 12;
    int k = kt * 32 + ((lane >> 4) * 8) + (j & 7) + ((j >> 3) * 16);
    int n = ct * 16 + (lane & 15);
    float v = (k < Kin) ? W[(size_t)k * HID + n] : 0.0f;
    dst[i] = f32_bf16(v);
}

// dst[node*Fout + f] = (f < Fin) ? src[node*Fin + f] : 0   (init agg = x, pad K)
__global__ void copy_pad_kernel(const float* __restrict__ src, int Fin,
                                float* __restrict__ dst, int Fout, int n) {
    long long i = (long long)blockIdx.x * blockDim.x + threadIdx.x;
    long long total = (long long)n * Fout;
    if (i >= total) return;
    int f = (int)(i % Fout);
    long long node = i / Fout;
    dst[i] = (f < Fin) ? src[node * Fin + f] : 0.0f;
}

// One wave per edge: agg[dst][f] += X[src][f]  (coalesced atomics)
__global__ void scatter_add_kernel(const float* __restrict__ X, int F, int strideX,
                                   const int* __restrict__ ei, int E,
                                   float* __restrict__ agg, int strideA) {
    int w = blockIdx.x * (blockDim.x >> 5) + (threadIdx.x >> 5);
    if (w >= E) return;
    int lane = threadIdx.x & 31;
    int s = ei[w];
    int d = ei[E + w];
    const float* xs = X + (size_t)s * strideX;
    float*       ad = agg + (size_t)d * strideA;
    for (int f = lane; f < F; f += 32)
        atomicAdd(&ad[f], xs[f]);
}

// C[16 nodes x 128] = act(A[16 x kpad] @ Wpack + bias) with v_wmma_f32_16x16x32_bf16.
// Block: 256 thr = 8 waves; wave ct owns columns ct*16..ct*16+15. grid.x = N/16.
__global__ void __launch_bounds__(256)
gemm_wmma_bf16_kernel(const float* __restrict__ A, int kpad,
                      const unsigned short* __restrict__ Wp,
                      const float* __restrict__ bias,
                      float* __restrict__ C, int relu) {
    __shared__ unsigned short lA[16 * 192];     // bf16 A tile, row-major [16][kpad]
    const int tid = threadIdx.x;
    const int blockRow = blockIdx.x * 16;

    // Stage + convert A tile f32 -> bf16 into LDS
    const int total = 16 * kpad;
    for (int i = tid; i < total; i += 256) {
        int row = i / kpad;
        int k   = i - row * kpad;
        lA[i] = f32_bf16(A[(size_t)(blockRow + row) * kpad + k]);
    }
    __syncthreads();

    const int lane = tid & 31;
    const int ct   = tid >> 5;
    const int nkt  = kpad >> 5;
    const int rowA = lane & 15;
    const int koff = (lane >> 4) * 8;

    v8f acc = {0.f, 0.f, 0.f, 0.f, 0.f, 0.f, 0.f, 0.f};
    union Frag { uint4 u[2]; v16bf v; };

    for (int kt = 0; kt < nkt; ++kt) {
        Frag a, b;
        // A fragment: 8 halves at koff, 8 halves at koff+16 (16B-aligned ds_load_b128 x2)
        const unsigned short* ap = &lA[rowA * kpad + kt * 32 + koff];
        a.u[0] = *(const uint4*)(ap);
        a.u[1] = *(const uint4*)(ap + 16);
        // B fragment: 32 contiguous bytes per lane from packed weights (global_load_b128 x2)
        const uint4* bp = (const uint4*)(Wp + (((size_t)kt * 8 + ct) * 32 + lane) * 16);
        b.u[0] = bp[0];
        b.u[1] = bp[1];
        acc = __builtin_amdgcn_wmma_f32_16x16x32_bf16(
                  false, a.v, false, b.v, (short)0, acc, false, false);
    }

    const int col   = ct * 16 + (lane & 15);
    const int rbase = (lane >> 4) * 8;
    const float bv  = bias[col];
    for (int r = 0; r < 8; ++r) {
        float v = acc[r] + bv;
        if (relu) v = fmaxf(v, 0.0f);
        C[(size_t)(blockRow + rbase + r) * HID + col] = v;
    }
}

__global__ void zero_kernel(float* p, int n) {
    int i = blockIdx.x * blockDim.x + threadIdx.x;
    if (i < n) p[i] = 0.0f;
}

// 128 threads/block, each block sums `perblk` nodes for its feature, scaled atomics -> mean
__global__ void pool_mean_kernel(const float* __restrict__ H, int n, int perblk,
                                 float* __restrict__ pool) {
    int f = threadIdx.x;
    int start = blockIdx.x * perblk;
    int end = min(start + perblk, n);
    float s = 0.0f;
    for (int node = start; node < end; ++node)
        s += H[(size_t)node * HID + f];
    atomicAdd(&pool[f], s * (1.0f / n));
}

// out = (pool @ Wc + bc) @ Wf + bf   -> logits [2]
__global__ void head_kernel(const float* __restrict__ pool,
                            const float* __restrict__ Wc, const float* __restrict__ bc,
                            const float* __restrict__ Wf, const float* __restrict__ bf,
                            float* __restrict__ out) {
    __shared__ float sp[HID];
    __shared__ float mid[HID];
    int t = threadIdx.x;
    sp[t] = pool[t];
    __syncthreads();
    float s = bc[t];
    for (int k = 0; k < HID; ++k) s += sp[k] * Wc[(size_t)k * HID + t];
    mid[t] = s;
    __syncthreads();
    if (t < 2) {
        float l = bf[t];
        for (int k = 0; k < HID; ++k) l += mid[k] * Wf[(size_t)k * 2 + t];
        out[t] = l;
    }
}

extern "C" void kernel_launch(void* const* d_in, const int* in_sizes, int n_in,
                              void* d_out, int out_size, void* d_ws, size_t ws_size,
                              hipStream_t stream) {
    const float* x   = (const float*)d_in[0];
    const int*   ei  = (const int*)d_in[1];
    const float* W1a = (const float*)d_in[2];  const float* b1a = (const float*)d_in[3];
    const float* W1b = (const float*)d_in[4];  const float* b1b = (const float*)d_in[5];
    const float* W2a = (const float*)d_in[6];  const float* b2a = (const float*)d_in[7];
    const float* W2b = (const float*)d_in[8];  const float* b2b = (const float*)d_in[9];
    const float* W3a = (const float*)d_in[10]; const float* b3a = (const float*)d_in[11];
    const float* W3b = (const float*)d_in[12]; const float* b3b = (const float*)d_in[13];
    const float* Wc  = (const float*)d_in[14]; const float* bc  = (const float*)d_in[15];
    const float* Wf  = (const float*)d_in[16]; const float* bf  = (const float*)d_in[17];

    // ---- carve workspace ----
    char* ws = (char*)d_ws;
    size_t off = 0;
    auto carve = [&](size_t bytes) -> void* {
        void* p = ws + off;
        off = (off + bytes + 255) & ~(size_t)255;
        return p;
    };
    float* AGG = (float*)carve((size_t)N_NODES * 192 * sizeof(float));
    float* T   = (float*)carve((size_t)N_NODES * HID * sizeof(float));
    float* H   = (float*)carve((size_t)N_NODES * HID * sizeof(float));
    unsigned short* P1a = (unsigned short*)carve((size_t)6 * 4096 * sizeof(unsigned short));
    unsigned short* P1b = (unsigned short*)carve((size_t)4 * 4096 * sizeof(unsigned short));
    unsigned short* P2a = (unsigned short*)carve((size_t)4 * 4096 * sizeof(unsigned short));
    unsigned short* P2b = (unsigned short*)carve((size_t)4 * 4096 * sizeof(unsigned short));
    unsigned short* P3a = (unsigned short*)carve((size_t)4 * 4096 * sizeof(unsigned short));
    unsigned short* P3b = (unsigned short*)carve((size_t)4 * 4096 * sizeof(unsigned short));
    float* pool = (float*)carve(HID * sizeof(float));

    // ---- pack weights into bf16 WMMA B fragments ----
    pack_w_kernel<<<(6 * 4096 + 255) / 256, 256, 0, stream>>>(W1a, P1a, 165, 6);
    pack_w_kernel<<<(4 * 4096 + 255) / 256, 256, 0, stream>>>(W1b, P1b, 128, 4);
    pack_w_kernel<<<(4 * 4096 + 255) / 256, 256, 0, stream>>>(W2a, P2a, 128, 4);
    pack_w_kernel<<<(4 * 4096 + 255) / 256, 256, 0, stream>>>(W2b, P2b, 128, 4);
    pack_w_kernel<<<(4 * 4096 + 255) / 256, 256, 0, stream>>>(W3a, P3a, 128, 4);
    pack_w_kernel<<<(4 * 4096 + 255) / 256, 256, 0, stream>>>(W3b, P3b, 128, 4);

    const int scatter_blocks = (N_EDGES + 7) / 8;   // 8 waves/block, 1 wave/edge
    const int gemm_blocks = N_NODES / 16;           // 6250, exact

    // ---- layer 1 (K = 165, padded to 192) ----
    {
        long long tot = (long long)N_NODES * 192;
        copy_pad_kernel<<<(unsigned)((tot + 255) / 256), 256, 0, stream>>>(x, 165, AGG, 192, N_NODES);
        scatter_add_kernel<<<scatter_blocks, 256, 0, stream>>>(x, 165, 165, ei, N_EDGES, AGG, 192);
        gemm_wmma_bf16_kernel<<<gemm_blocks, 256, 0, stream>>>(AGG, 192, P1a, b1a, T, 1);
        gemm_wmma_bf16_kernel<<<gemm_blocks, 256, 0, stream>>>(T, 128, P1b, b1b, H, 1);
    }
    // ---- layers 2 and 3 (K = 128) ----
    const unsigned short* Pa[2] = {P2a, P3a};
    const unsigned short* Pb[2] = {P2b, P3b};
    const float* Ba[2] = {b2a, b3a};
    const float* Bb[2] = {b2b, b3b};
    for (int layer = 0; layer < 2; ++layer) {
        long long tot = (long long)N_NODES * 128;
        copy_pad_kernel<<<(unsigned)((tot + 255) / 256), 256, 0, stream>>>(H, 128, AGG, 128, N_NODES);
        scatter_add_kernel<<<scatter_blocks, 256, 0, stream>>>(H, 128, 128, ei, N_EDGES, AGG, 128);
        gemm_wmma_bf16_kernel<<<gemm_blocks, 256, 0, stream>>>(AGG, 128, Pa[layer], Ba[layer], T, 1);
        gemm_wmma_bf16_kernel<<<gemm_blocks, 256, 0, stream>>>(T, 128, Pb[layer], Bb[layer], H, 1);
    }

    // ---- global mean pool + head ----
    zero_kernel<<<1, 128, 0, stream>>>(pool, HID);
    const int perblk = 500;                         // 200 blocks cover 100000 nodes
    pool_mean_kernel<<<(N_NODES + perblk - 1) / perblk, 128, 0, stream>>>(H, N_NODES, perblk, pool);
    head_kernel<<<1, 128, 0, stream>>>(pool, Wc, bc, Wf, bf, (float*)d_out);

    // ---- second tuple element: pass edge_index through ----
    size_t avail = (out_size > 2) ? (size_t)(out_size - 2) * sizeof(float) : 0;
    size_t want  = (size_t)2 * N_EDGES * sizeof(int);
    size_t nbytes = avail < want ? avail : want;
    if (nbytes)
        hipMemcpyAsync((float*)d_out + 2, d_in[1], nbytes, hipMemcpyDeviceToDevice, stream);
}
